// MPNN_64845416235323
// MI455X (gfx1250) — matrix operand order repaired and checked by reading, not compiled, not run on hardware
//
#include <hip/hip_runtime.h>
#include <hip/hip_bf16.h>

// ---- problem constants ----
#define NN   68           // nodes
#define BB   64           // batch
#define DD   256          // feature dim
#define TT   3            // iterations
#define MM   (BB*NN)      // 4352 rows, = 272 * 16
#define MT   (MM/16)      // 272 M-tiles
#define NEGS 0.01f
#define EPSL 1e-5f

typedef __attribute__((ext_vector_type(16))) __bf16 v16bf;
typedef __attribute__((ext_vector_type(8)))  __bf16 bf16x8;
typedef __attribute__((ext_vector_type(8)))  float  v8f;

static __device__ __forceinline__ __bf16 f2bf(float f) {
  unsigned u = __float_as_uint(f);
  u += 0x7FFFu + ((u >> 16) & 1u);              // round-to-nearest-even
  unsigned short h = (unsigned short)(u >> 16);
  return __builtin_bit_cast(__bf16, h);
}

static __device__ __forceinline__ float lrelu(float v) {
  return v > 0.f ? v : NEGS * v;
}

// ---------------- targ broadcast: d_out[T*M .. 2*T*M) = target tiled ----------
__global__ void fill_targ_kernel(const float* __restrict__ target, float* __restrict__ out) {
  int id = blockIdx.x * blockDim.x + threadIdx.x;
  if (id < TT * MM) out[id] = target[id % MM];
}

// ---------------- weight prep: f32 -> bf16, row-major (Nout x K) --------------
// Wcat rows 0..255  = Wj  = W_mp[n, 0:256]
// Wcat rows 256..511 = Wi = W_mp[n-256, 256:512]
__global__ void prep_weights_kernel(const float* __restrict__ W_mp, const float* __restrict__ W_up,
                                    __bf16* __restrict__ Wcat, __bf16* __restrict__ Wupb) {
  int id = blockIdx.x * blockDim.x + threadIdx.x;
  if (id < 512 * 256) {
    int n = id >> 8, k = id & 255;
    float v = (n < 256) ? W_mp[n * 516 + k] : W_mp[(n - 256) * 516 + 256 + k];
    Wcat[id] = f2bf(v);
  } else {
    int id2 = id - 512 * 256;
    if (id2 < 256 * 256) Wupb[id2] = f2bf(W_up[id2]);
  }
}

__global__ void conv_bf16_kernel(const float* __restrict__ src, __bf16* __restrict__ dst, int n) {
  int id = blockIdx.x * blockDim.x + threadIdx.x;
  if (id < n) dst[id] = f2bf(src[id]);
}

// ---------------- WMMA GEMM: Out[M x (Ntiles*16)] = A[M x 256] * Bw^T ---------
// A row-major bf16 (ld=256), Bw row-major bf16 (Nout x 256): Out[m,n]=sum_k A[m,k]*Bw[n,k]
// one wave per 16x16 output tile; K-loop of 8 x v_wmma_f32_16x16x32_bf16
__global__ void gemm_bf16_kernel(const __bf16* __restrict__ A, const __bf16* __restrict__ Bw,
                                 float* __restrict__ Out, int Ntiles) {
  const int K = DD;
  int wave = blockIdx.x * (blockDim.x >> 5) + (threadIdx.x >> 5);
  int lane = threadIdx.x & 31;
  int tN = wave % Ntiles, tM = wave / Ntiles;
  int half = lane >> 4, r16 = lane & 15;

  const __bf16* arow = A  + (size_t)(tM * 16 + r16) * K;
  const __bf16* brow = Bw + (size_t)(tN * 16 + r16) * K;
  v8f c = {};
#pragma unroll
  for (int kk = 0; kk < K; kk += 32) {
    bf16x8 alo = *(const bf16x8*)(arow + kk + half * 16);       // K = kk + half*16 + [0..7]
    bf16x8 ahi = *(const bf16x8*)(arow + kk + half * 16 + 8);   // K = kk + half*16 + 8 + [0..7]
    v16bf b = *(const v16bf*)(brow + kk + half * 16);           // K = kk + half*16 + [0..15]
    v16bf a;
#pragma unroll
    for (int i = 0; i < 8; ++i) { a[i] = alo[i]; a[8 + i] = ahi[i]; }
    c = __builtin_amdgcn_wmma_f32_16x16x32_bf16(false, a, false, b, (short)0, c, false, false);
  }
  int ldO = Ntiles * 16;
  float* orow = Out + (size_t)(tM * 16 + 8 * half) * ldO + tN * 16 + r16;
#pragma unroll
  for (int r = 0; r < 8; ++r) orow[(size_t)r * ldO] = c[r];     // D: M = r + 8*half, N = r16
}

// ---------------- aggregation: m = H + (1/N) * sum_{j!=i} lrelu(Aj+Bi+Ee) -----
// AB layout: [M x 512] f32, cols 0..255 = Aj, 256..511 = Bi.  Ee recomputed on the fly.
__global__ void agg_kernel(const float* __restrict__ AB, const float* __restrict__ Hf,
                           const float* __restrict__ E, const float* __restrict__ W_mp,
                           const float* __restrict__ b_mp, __bf16* __restrict__ Mb) {
  __shared__ float e_lds[4 * NN];
  int bi_idx = blockIdx.x;             // b*N + i
  int b = bi_idx / NN, i = bi_idx % NN;
  int k = threadIdx.x;                 // feature 0..255

  for (int t = threadIdx.x; t < 4 * NN; t += blockDim.x) {
    int f = t / NN, j = t % NN;
    e_lds[t] = E[f * NN * NN + i * NN + j];
  }
  __syncthreads();

  float we0 = W_mp[k * 516 + 512], we1 = W_mp[k * 516 + 513];
  float we2 = W_mp[k * 516 + 514], we3 = W_mp[k * 516 + 515];
  float base = AB[(size_t)bi_idx * 512 + 256 + k] + b_mp[k];   // Bi[b,i,k] + b_mp[k]
  const float* ajp = AB + (size_t)b * NN * 512 + k;

  float acc = 0.f;
#pragma unroll 4
  for (int j = 0; j < NN; ++j) {
    if (j == i) continue;
    float e = we0 * e_lds[j] + we1 * e_lds[NN + j] + we2 * e_lds[2 * NN + j] + we3 * e_lds[3 * NN + j];
    acc += lrelu(ajp[(size_t)j * 512] + base + e);
  }
  float m = Hf[(size_t)bi_idx * DD + k] + acc * (1.0f / (float)NN);
  Mb[(size_t)bi_idx * DD + k] = f2bf(m);
}

// ---------------- fused update: u=lrelu(m*Wup^T+b), layernorm, readout --------
// one block = 16 rows x 256 cols; 8 waves, wave w owns column tiles 2w, 2w+1
__global__ void update_kernel(const __bf16* __restrict__ Mb, const __bf16* __restrict__ Wupb,
                              const float* __restrict__ b_up, const float* __restrict__ W_ro,
                              const float* __restrict__ b_ro,
                              float* __restrict__ Hn, __bf16* __restrict__ Hnb,
                              float* __restrict__ pred) {
  __shared__ float u_lds[16 * DD];
  const int K = DD;
  int tM = blockIdx.x;
  int wave = threadIdx.x >> 5, lane = threadIdx.x & 31;
  int half = lane >> 4, r16 = lane & 15;
  int n0 = wave * 2, n1 = wave * 2 + 1;

  const __bf16* arow  = Mb   + (size_t)(tM * 16 + r16) * K;
  const __bf16* b0row = Wupb + (size_t)(n0 * 16 + r16) * K;
  const __bf16* b1row = Wupb + (size_t)(n1 * 16 + r16) * K;
  v8f c0 = {}, c1 = {};
#pragma unroll
  for (int kk = 0; kk < K; kk += 32) {
    bf16x8 alo = *(const bf16x8*)(arow + kk + half * 16);
    bf16x8 ahi = *(const bf16x8*)(arow + kk + half * 16 + 8);
    v16bf a;
#pragma unroll
    for (int i = 0; i < 8; ++i) { a[i] = alo[i]; a[8 + i] = ahi[i]; }
    v16bf b0 = *(const v16bf*)(b0row + kk + half * 16);
    v16bf b1 = *(const v16bf*)(b1row + kk + half * 16);
    c0 = __builtin_amdgcn_wmma_f32_16x16x32_bf16(false, a, false, b0, (short)0, c0, false, false);
    c1 = __builtin_amdgcn_wmma_f32_16x16x32_bf16(false, a, false, b1, (short)0, c1, false, false);
  }
  float bias0 = b_up[n0 * 16 + r16], bias1 = b_up[n1 * 16 + r16];
#pragma unroll
  for (int r = 0; r < 8; ++r) {
    int row = r + 8 * half;
    u_lds[row * DD + n0 * 16 + r16] = lrelu(c0[r] + bias0);
    u_lds[row * DD + n1 * 16 + r16] = lrelu(c1[r] + bias1);
  }
  __syncthreads();

  // layernorm + readout: 16 lanes per row, wave32 shfl_xor reductions
  int row = threadIdx.x >> 4, l16 = threadIdx.x & 15;
  float s = 0.f, q = 0.f;
#pragma unroll
  for (int ii = 0; ii < 16; ++ii) {
    float v = u_lds[row * DD + l16 + 16 * ii];
    s += v; q += v * v;
  }
#pragma unroll
  for (int off = 8; off >= 1; off >>= 1) {
    s += __shfl_xor(s, off, 32);
    q += __shfl_xor(q, off, 32);
  }
  float mean = s * (1.f / (float)DD);
  float var  = q * (1.f / (float)DD) - mean * mean;
  float inv  = rsqrtf(var + EPSL);

  size_t grow = (size_t)tM * 16 + row;
  float dot = 0.f;
#pragma unroll
  for (int ii = 0; ii < 16; ++ii) {
    int col = l16 + 16 * ii;
    float hn = (u_lds[row * DD + col] - mean) * inv;
    Hn[grow * DD + col]  = hn;
    Hnb[grow * DD + col] = f2bf(hn);
    dot += hn * W_ro[col];
  }
#pragma unroll
  for (int off = 8; off >= 1; off >>= 1) dot += __shfl_xor(dot, off, 32);
  if (l16 == 0) {
    float z = dot + b_ro[0];
    pred[grow] = 1.0f / (1.0f + __expf(-z));
  }
}

// -----------------------------------------------------------------------------
extern "C" void kernel_launch(void* const* d_in, const int* in_sizes, int n_in,
                              void* d_out, int out_size, void* d_ws, size_t ws_size,
                              hipStream_t stream) {
  (void)in_sizes; (void)n_in; (void)out_size; (void)ws_size;
  const float* latent = (const float*)d_in[0];   // (64,68,256)
  const float* target = (const float*)d_in[1];   // (64,68)
  const float* E      = (const float*)d_in[2];   // (4,68,68)
  const float* W_mp   = (const float*)d_in[3];   // (256,516)
  const float* b_mp   = (const float*)d_in[4];   // (256,)
  const float* W_up   = (const float*)d_in[5];   // (256,256)
  const float* b_up   = (const float*)d_in[6];   // (256,)
  const float* W_ro   = (const float*)d_in[7];   // (1,256)
  const float* b_ro   = (const float*)d_in[8];   // (1,)
  float* out = (float*)d_out;                    // pred (3,64,68) ++ targ (3,64,68)

  char* ws = (char*)d_ws;
  size_t off = 0;
  auto wsalloc = [&](size_t bytes) -> void* {
    void* p = ws + off;
    off += (bytes + 255) & ~(size_t)255;
    return p;
  };
  float*  AB   = (float*)  wsalloc((size_t)MM * 512 * sizeof(float)); // Aj|Bi
  float*  Hf   = (float*)  wsalloc((size_t)MM * DD * sizeof(float));  // Hn f32
  __bf16* Hb   = (__bf16*) wsalloc((size_t)MM * DD * 2);              // H bf16 (GEMM A)
  __bf16* Mb   = (__bf16*) wsalloc((size_t)MM * DD * 2);              // m bf16
  __bf16* Wcat = (__bf16*) wsalloc((size_t)512 * DD * 2);             // [Wj;Wi] rows
  __bf16* Wupb = (__bf16*) wsalloc((size_t)DD * DD * 2);              // W_up rows

  fill_targ_kernel<<<(TT * MM + 255) / 256, 256, 0, stream>>>(target, out + TT * MM);
  prep_weights_kernel<<<(512 * 256 + 256 * 256) / 256, 256, 0, stream>>>(W_mp, W_up, Wcat, Wupb);
  conv_bf16_kernel<<<(MM * DD + 255) / 256, 256, 0, stream>>>(latent, Hb, MM * DD);

  const float* Hcur = latent;
  for (int t = 0; t < TT; ++t) {
    // Aj|Bi GEMM: 272 x 32 tiles, 8 waves/block
    gemm_bf16_kernel<<<(MT * 32) / 8, 256, 0, stream>>>(Hb, Wcat, AB, 32);
    // aggregation (Ee folded in)
    agg_kernel<<<MM, 256, 0, stream>>>(AB, Hcur, E, W_mp, b_mp, Mb);
    // fused update GEMM + layernorm + readout; refreshes Hf (f32) and Hb (bf16)
    update_kernel<<<MT, 256, 0, stream>>>(Mb, Wupb, b_up, W_ro, b_ro, Hf, Hb, out + t * MM);
    Hcur = Hf;
  }
}
